// WSTOG_81552839016613
// MI455X (gfx1250) — compile-verified, matrix operand order checked by reading, not run on hardware
//
#include <hip/hip_runtime.h>
#include <hip/hip_bf16.h>
#include <math.h>

// ---------------- problem constants ----------------
constexpr int Bb = 20, Tt = 20, Rn = 20, Dd = 4096, Oo = 16, Ee = 512,
              Hh = 8, TMm = 1900;
constexpr int Nkv = Tt * Rn + Tt;   // 420 (regions + retrieved memory)
constexpr int Dh  = Ee / Hh;        // 64
constexpr int NHI = Bb * Oo * Tt;   // 6400 hi rows (lo rows follow)

// ---------------- CDNA5 WMMA vector types ----------------
typedef __attribute__((ext_vector_type(16))) __bf16   v16bf;
typedef __attribute__((ext_vector_type(8)))  float    v8f;
typedef __attribute__((ext_vector_type(8)))  unsigned v8u;

__device__ __forceinline__ unsigned short f2bf(float f) {
  unsigned u = __builtin_bit_cast(unsigned, f);
  u += 0x7FFFu + ((u >> 16) & 1u);       // round-to-nearest-even
  return (unsigned short)(u >> 16);
}

// ---------------- block reductions (wave32-aware) ----------------
__device__ __forceinline__ float block_reduce_sum(float v) {
  __shared__ float red[8];
  int lane = threadIdx.x & 31, wid = threadIdx.x >> 5;
#pragma unroll
  for (int o = 16; o > 0; o >>= 1) v += __shfl_xor(v, o, 32);
  if (lane == 0) red[wid] = v;
  __syncthreads();
  int nw = (blockDim.x + 31) >> 5;
  if (wid == 0) {
    float s = (lane < nw) ? red[lane] : 0.f;
#pragma unroll
    for (int o = 4; o > 0; o >>= 1) s += __shfl_xor(s, o, 32);
    if (lane == 0) red[0] = s;
  }
  __syncthreads();
  float r = red[0];
  __syncthreads();
  return r;
}

__device__ __forceinline__ float block_reduce_max(float v) {
  __shared__ float red[8];
  int lane = threadIdx.x & 31, wid = threadIdx.x >> 5;
#pragma unroll
  for (int o = 16; o > 0; o >>= 1) v = fmaxf(v, __shfl_xor(v, o, 32));
  if (lane == 0) red[wid] = v;
  __syncthreads();
  int nw = (blockDim.x + 31) >> 5;
  if (wid == 0) {
    float s = (lane < nw) ? red[lane] : -1e30f;
#pragma unroll
    for (int o = 4; o > 0; o >>= 1) s = fmaxf(s, __shfl_xor(s, o, 32));
    if (lane == 0) red[0] = s;
  }
  __syncthreads();
  float r = red[0];
  __syncthreads();
  return r;
}

// ---------------- generic bf16-WMMA GEMM ----------------
// C[M,N] = act( A[M,K] * B + bias ),  A row-major (optionally row-gathered),
// B either [K,N] row-major (BT=false) or [N,K] row-major (BT=true).
// EPI: 0 = none, 1 = tanh(x + bias[n]).
template <int EPI, bool BT, bool RM>
__global__ __launch_bounds__(256) void k_gemm_bf16(
    const float* __restrict__ A, const float* __restrict__ Bm,
    float* __restrict__ Cm, const int* __restrict__ rowmap,
    const float* __restrict__ bias, int M, int N, int K, int lda, int ldb,
    int ldc) {
  __shared__ unsigned short As[64][34];  // [m][k]  (pad keeps u32 aligned)
  __shared__ unsigned short Bs[64][40];  // [n][k]
  const int t  = threadIdx.x;
  const int m0 = blockIdx.y * 64;
  const int n0 = blockIdx.x * 64;

  const int wv = t >> 5, lane = t & 31;
  const int half = lane >> 4, mr = lane & 15;
  const int sr = wv & 3;            // subtile row 0..3
  const int scb = (wv >> 2) << 1;   // subtile col pair base {0,2}

  v8f acc0 = {}, acc1 = {};

  // A tile loader: 4 threads per row, 8 consecutive k each.
  const int arow = t >> 2, acb = (t & 3) << 3;
  int gm = m0 + arow;
  const bool mok = gm < M;
  const float* arowp = nullptr;
  if (mok) {
    int rr = RM ? rowmap[gm] : gm;
    arowp = A + (size_t)rr * lda + acb;
  }

  for (int k0 = 0; k0 < K; k0 += 32) {
    // ---- stage A (f32 -> bf16) ----
    {
      float va[8];
      if (mok) {
        const float* p = arowp + k0;
#pragma unroll
        for (int j = 0; j < 8; ++j) va[j] = p[j];
        if (k0 + 32 < K) __builtin_prefetch(p + 32, 0, 0);  // global_prefetch_b8
      } else {
#pragma unroll
        for (int j = 0; j < 8; ++j) va[j] = 0.f;
      }
#pragma unroll
      for (int j = 0; j < 8; ++j) As[arow][acb + j] = f2bf(va[j]);
    }
    // ---- stage B into Bs[n][k] ----
    if (!BT) {
      const int br = t >> 3, bcb = (t & 7) << 3;  // k row, n col base
#pragma unroll
      for (int j = 0; j < 8; ++j) {
        int n = n0 + bcb + j;
        float v = (n < N) ? Bm[(size_t)(k0 + br) * ldb + n] : 0.f;
        Bs[bcb + j][br] = f2bf(v);
      }
    } else {
      const int bn = t >> 2, bkb = (t & 3) << 3;  // n row, k col base
      int n = n0 + bn;
#pragma unroll
      for (int j = 0; j < 8; ++j) {
        float v = (n < N) ? Bm[(size_t)n * ldb + k0 + bkb + j] : 0.f;
        Bs[bn][bkb + j] = f2bf(v);
      }
    }
    __syncthreads();

    // ---- fragments per CDNA5 16-bit striping: k(j) = (j>=8? j+8 : j) + half*8
    v8u au, bu0, bu1;
#pragma unroll
    for (int p = 0; p < 8; ++p) {
      int kk = ((p >> 2) << 4) + ((p & 3) << 1) + (half << 3);
      au[p]  = *(const unsigned*)&As[sr * 16 + mr][kk];
      bu0[p] = *(const unsigned*)&Bs[scb * 16 + mr][kk];
      bu1[p] = *(const unsigned*)&Bs[(scb + 1) * 16 + mr][kk];
    }
    v16bf a  = __builtin_bit_cast(v16bf, au);
    v16bf b0 = __builtin_bit_cast(v16bf, bu0);
    v16bf b1 = __builtin_bit_cast(v16bf, bu1);
    acc0 = __builtin_amdgcn_wmma_f32_16x16x32_bf16(false, a, false, b0,
                                                   (short)0, acc0, false, false);
    acc1 = __builtin_amdgcn_wmma_f32_16x16x32_bf16(false, a, false, b1,
                                                   (short)0, acc1, false, false);
    __syncthreads();
  }

  // ---- store per documented C/D layout ----
#pragma unroll
  for (int v = 0; v < 8; ++v) {
    int m = m0 + sr * 16 + half * 8 + v;
    if (m >= M) continue;
    int na = n0 + scb * 16 + mr;
    int nb = na + 16;
    if (na < N) {
      float x = acc0[v];
      if (EPI == 1) x = tanhf(x + bias[na]);
      Cm[(size_t)m * ldc + na] = x;
    }
    if (nb < N) {
      float x = acc1[v];
      if (EPI == 1) x = tanhf(x + bias[nb]);
      Cm[(size_t)m * ldc + nb] = x;
    }
  }
}

// ---------------- streaming / pointwise kernels ----------------

// l2-normalize each feature row into KV[b, t*Rn+r, :]
__global__ void k_norm_features(const float* __restrict__ F,
                                float* __restrict__ KV) {
  int row = blockIdx.x;                       // b*T*Rn + t*Rn + r
  int b = row / (Tt * Rn), tr = row % (Tt * Rn);
  const float4* src = (const float4*)(F + (size_t)row * Dd);
  float ss = 0.f;
  for (int i = threadIdx.x; i < Dd / 4; i += blockDim.x) {
    float4 v = src[i];
    ss += v.x * v.x + v.y * v.y + v.z * v.z + v.w * v.w;
  }
  ss = block_reduce_sum(ss);
  float rn = 1.f / fmaxf(sqrtf(ss), 1e-12f);
  float4* dst = (float4*)(KV + ((size_t)b * Nkv + tr) * Dd);
  for (int i = threadIdx.x; i < Dd / 4; i += blockDim.x) {
    float4 v = src[i];
    v.x *= rn; v.y *= rn; v.z *= rn; v.w *= rn;
    dst[i] = v;
  }
}

// R_max = l2norm(max over regions of Rf)
__global__ void k_rmax(const float* __restrict__ KV, float* __restrict__ RM) {
  int bt = blockIdx.x;
  int b = bt / Tt, t = bt % Tt;
  const float* base = KV + ((size_t)b * Nkv + (size_t)t * Rn) * Dd;
  float mx[Dd / 256];
  float ss = 0.f;
#pragma unroll
  for (int j = 0; j < Dd / 256; ++j) {
    int d = threadIdx.x + j * 256;
    float m = -1e30f;
    for (int r = 0; r < Rn; ++r) m = fmaxf(m, base[(size_t)r * Dd + d]);
    mx[j] = m;
    ss += m * m;
  }
  ss = block_reduce_sum(ss);
  float rn = 1.f / fmaxf(sqrtf(ss), 1e-12f);
#pragma unroll
  for (int j = 0; j < Dd / 256; ++j)
    RM[(size_t)bt * Dd + threadIdx.x + j * 256] = mx[j] * rn;
}

// mem_max = l2norm(max over frames of temporal_mem)
__global__ void k_memmax(const float* __restrict__ TMEM,
                         float* __restrict__ MM) {
  int m = blockIdx.x;
  const float* base = TMEM + (size_t)m * Tt * Dd;
  float mx[Dd / 256];
  float ss = 0.f;
#pragma unroll
  for (int j = 0; j < Dd / 256; ++j) {
    int d = threadIdx.x + j * 256;
    float v = -1e30f;
    for (int f = 0; f < Tt; ++f) v = fmaxf(v, base[(size_t)f * Dd + d]);
    mx[j] = v;
    ss += v * v;
  }
  ss = block_reduce_sum(ss);
  float rn = 1.f / fmaxf(sqrtf(ss), 1e-12f);
#pragma unroll
  for (int j = 0; j < Dd / 256; ++j)
    MM[(size_t)m * Dd + threadIdx.x + j * 256] = mx[j] * rn;
}

// row-wise argmax over TM (first occurrence wins, like jnp.argmax)
__global__ void k_argmax(const float* __restrict__ SIM, int* __restrict__ IDX) {
  __shared__ float sv[256];
  __shared__ int si[256];
  int row = blockIdx.x;
  const float* s = SIM + (size_t)row * TMm;
  float best = -1e30f;
  int bi = 0;
  for (int i = threadIdx.x; i < TMm; i += blockDim.x) {
    float v = s[i];
    if (v > best) { best = v; bi = i; }
  }
  sv[threadIdx.x] = best;
  si[threadIdx.x] = bi;
  __syncthreads();
  for (int off = 128; off > 0; off >>= 1) {
    if (threadIdx.x < off) {
      float v2 = sv[threadIdx.x + off];
      int i2 = si[threadIdx.x + off];
      if (v2 > sv[threadIdx.x] ||
          (v2 == sv[threadIdx.x] && i2 < si[threadIdx.x])) {
        sv[threadIdx.x] = v2;
        si[threadIdx.x] = i2;
      }
    }
    __syncthreads();
  }
  if (threadIdx.x == 0) IDX[row] = si[0];
}

// mem_feat[b,t,:] = max over frames of temporal_mem[idx[b,t]] -> KV rows 400..419
__global__ void k_memfeat(const float* __restrict__ TMEM,
                          const int* __restrict__ IDX, float* __restrict__ KV) {
  int bt = blockIdx.x;
  int b = bt / Tt, t = bt % Tt;
  const float* src = TMEM + (size_t)IDX[bt] * Tt * Dd;
  float* dst = KV + ((size_t)b * Nkv + Tt * Rn + t) * Dd;
  for (int d = threadIdx.x; d < Dd; d += blockDim.x) {
    float mx = -1e30f;
    for (int f = 0; f < Tt; ++f) mx = fmaxf(mx, src[(size_t)f * Dd + d]);
    dst[d] = mx;
  }
}

// Qe = (labels!=0) * entity_W[labels] + entity_b
__global__ void k_make_qe(const int* __restrict__ labels,
                          const float* __restrict__ EW,
                          const float* __restrict__ Eb,
                          float* __restrict__ QE) {
  int i = blockIdx.x;
  int lab = labels[i];
  float msk = (lab != 0) ? 1.f : 0.f;
  const float* w = EW + (size_t)lab * Ee;
  float* dst = QE + (size_t)i * Ee;
  for (int c = threadIdx.x; c < Ee; c += blockDim.x)
    dst[c] = msk * w[c] + Eb[c];
}

// scores[b,h,o,n] = (q . k) / sqrt(Dh)
__global__ void k_scores(const float* __restrict__ Q,
                         const float* __restrict__ Kf,
                         float* __restrict__ SC) {
  int blk = blockIdx.x;  // ((b*H)+h)*O + o
  int b = blk / (Hh * Oo);
  int rem = blk % (Hh * Oo);
  int h = rem / Oo, o = rem % Oo;
  const float* qrow = Q + ((size_t)b * Oo + o) * Ee + h * Dh;
  for (int n = threadIdx.x; n < Nkv; n += blockDim.x) {
    const float* krow = Kf + ((size_t)b * Nkv + n) * Ee + h * Dh;
    float s = 0.f;
#pragma unroll 8
    for (int d = 0; d < Dh; ++d) s += qrow[d] * krow[d];
    SC[(size_t)blk * Nkv + n] = s * 0.125f;  // 1/sqrt(64)
  }
}

// softmax over n + attn @ v  ->  AOUT[b*O+o, h*Dh+d]
__global__ void k_attn(const float* __restrict__ SC,
                       const float* __restrict__ V, float* __restrict__ AOUT) {
  __shared__ float p[Nkv];
  int blk = blockIdx.x;
  int b = blk / (Hh * Oo);
  int rem = blk % (Hh * Oo);
  int h = rem / Oo, o = rem % Oo;
  const float* sr = SC + (size_t)blk * Nkv;
  float mx = -1e30f;
  for (int n = threadIdx.x; n < Nkv; n += blockDim.x) {
    float v = sr[n];
    p[n] = v;
    mx = fmaxf(mx, v);
  }
  mx = block_reduce_max(mx);
  float sum = 0.f;
  for (int n = threadIdx.x; n < Nkv; n += blockDim.x) {
    float e = __expf(p[n] - mx);
    p[n] = e;
    sum += e;
  }
  sum = block_reduce_sum(sum);
  float inv = 1.f / sum;
  int d = threadIdx.x;
  if (d < Dh) {
    const float* vb = V + (size_t)b * Nkv * Ee + h * Dh + d;
    float acc = 0.f;
    for (int n = 0; n < Nkv; ++n) acc += p[n] * vb[(size_t)n * Ee];
    AOUT[((size_t)b * Oo + o) * Ee + h * Dh + d] = acc * inv;
  }
}

// head-mean of scores, argmax/argmin over regions -> KV row map (hi || lo)
__global__ void k_qr(const float* __restrict__ SC, int* __restrict__ rowmap) {
  int bo = blockIdx.x;
  int b = bo / Oo, o = bo % Oo;
  int t = threadIdx.x;
  if (t >= Tt) return;
  float best = -1e30f, worst = 1e30f;
  int hi = 0, lo = 0;
  for (int r = 0; r < Rn; ++r) {
    float s = 0.f;
    for (int h = 0; h < Hh; ++h)
      s += SC[(((size_t)(b * Hh + h)) * Oo + o) * Nkv + t * Rn + r];
    s *= (1.f / Hh);
    if (s > best) { best = s; hi = r; }
    if (s < worst) { worst = s; lo = r; }
  }
  int i = bo * Tt + t;
  rowmap[i]       = b * Nkv + t * Rn + hi;
  rowmap[NHI + i] = b * Nkv + t * Rn + lo;
}

// s = sigmoid(z . W2 + b2)
__global__ void k_mlp2(const float* __restrict__ Z, const float* __restrict__ W2,
                       const float* __restrict__ b2, float* __restrict__ S) {
  int row = blockIdx.x;
  const float* z = Z + (size_t)row * Ee;
  float acc = 0.f;
  for (int i = threadIdx.x; i < Ee; i += blockDim.x) acc += z[i] * W2[i];
  acc = block_reduce_sum(acc);
  if (threadIdx.x == 0) S[row] = 1.f / (1.f + __expf(-(acc + b2[0])));
}

// generalized-CE reduction to a scalar
__global__ void k_gce(const float* __restrict__ S, const int* __restrict__ labels,
                      float* __restrict__ G) {
  __shared__ float shi[Bb], slo[Bb];
  if (threadIdx.x < Bb) { shi[threadIdx.x] = 0.f; slo[threadIdx.x] = 0.f; }
  __syncthreads();
  const float q = 0.7f;
  for (int i = threadIdx.x; i < NHI; i += blockDim.x) {
    int b = i / (Oo * Tt);
    int o = (i / Tt) % Oo;
    float osel = (labels[b * Oo + o] != 0) ? 1.f : 0.f;
    float sh = S[i], sl = S[NHI + i];
    float thi = (1.f - powf(sh + 1e-7f, q)) / q * osel * (1.f / Tt);
    float tlo = (1.f - powf(1.f - sl + 1e-7f, q)) / q * osel * (1.f / Tt);
    atomicAdd(&shi[b], thi);
    atomicAdd(&slo[b], tlo);
  }
  __syncthreads();
  if (threadIdx.x == 0) {
    float g = 0.f;
    for (int b = 0; b < Bb; ++b) {
      float on = 0.f;
      for (int o = 0; o < Oo; ++o) on += (labels[b * Oo + o] != 0) ? 1.f : 0.f;
      on = fmaxf(on, 1.f);
      g += shi[b] / on + slo[b] / on;
    }
    G[0] = g / (float)Bb;
  }
}

__global__ void k_final(const float* __restrict__ FUSED,
                        const float* __restrict__ G, float* __restrict__ out,
                        int n) {
  int i = blockIdx.x * blockDim.x + threadIdx.x;
  if (i < n) out[i] = FUSED[i] + 0.5f * G[0];
}

// ---------------- host launch ----------------
static inline dim3 gemm_grid(int M, int N) {
  return dim3((N + 63) / 64, (M + 63) / 64);
}

extern "C" void kernel_launch(void* const* d_in, const int* in_sizes, int n_in,
                              void* d_out, int out_size, void* d_ws,
                              size_t ws_size, hipStream_t stream) {
  const float* features   = (const float*)d_in[0];
  const int*   labels     = (const int*)d_in[1];
  const float* tmem       = (const float*)d_in[2];
  const float* entity_W   = (const float*)d_in[3];
  const float* entity_b   = (const float*)d_in[4];
  const float* Wq         = (const float*)d_in[5];
  const float* Wk         = (const float*)d_in[6];
  const float* Wv         = (const float*)d_in[7];
  const float* Wo         = (const float*)d_in[8];
  const float* obj_W1     = (const float*)d_in[9];
  const float* obj_b1     = (const float*)d_in[10];
  const float* obj_W2     = (const float*)d_in[11];
  const float* obj_b2     = (const float*)d_in[12];
  float* out = (float*)d_out;

  float* ws = (float*)d_ws;
  size_t off = 0;
  auto alloc = [&](size_t n) {
    size_t o = off;
    off += (n + 255) & ~(size_t)255;
    return o;
  };
  float* KV     = ws + alloc((size_t)Bb * Nkv * Dd);   // 34.4M (Rf + mem_feat)
  float* RMAX   = ws + alloc((size_t)Bb * Tt * Dd);
  float* MEMMAX = ws + alloc((size_t)TMm * Dd);
  float* SIM    = ws + alloc((size_t)Bb * Tt * TMm);
  float* QE     = ws + alloc((size_t)Bb * Oo * Ee);
  float* QBUF   = ws + alloc((size_t)Bb * Oo * Ee);
  float* KBUF   = ws + alloc((size_t)Bb * Nkv * Ee);
  float* VBUF   = ws + alloc((size_t)Bb * Nkv * Ee);
  float* SCORES = ws + alloc((size_t)Bb * Hh * Oo * Nkv);
  float* AOUT   = ws + alloc((size_t)Bb * Oo * Ee);
  float* ZBUF   = ws + alloc((size_t)2 * NHI * Ee);
  float* SBUF   = ws + alloc((size_t)2 * NHI);
  float* FUSED  = ws + alloc((size_t)Bb * Oo * Ee);
  float* GCE    = ws + alloc(256);
  int* IDX      = (int*)(ws + alloc(512));
  int* ROWMAP   = (int*)(ws + alloc(2 * NHI));
  (void)ws_size; (void)in_sizes; (void)n_in; (void)out_size;

  // 1) l2norm features -> KV[:, 0:400, :]
  k_norm_features<<<Bb * Tt * Rn, 256, 0, stream>>>(features, KV);
  // 2) pooled+normalized query / memory banks
  k_rmax<<<Bb * Tt, 256, 0, stream>>>(KV, RMAX);
  k_memmax<<<TMm, 256, 0, stream>>>(tmem, MEMMAX);
  // 3) sim = R_max @ mem_max^T   (WMMA, B transposed)
  k_gemm_bf16<0, true, false><<<gemm_grid(Bb * Tt, TMm), 256, 0, stream>>>(
      RMAX, MEMMAX, SIM, nullptr, nullptr, Bb * Tt, TMm, Dd, Dd, Dd, TMm);
  k_argmax<<<Bb * Tt, 256, 0, stream>>>(SIM, IDX);
  k_memfeat<<<Bb * Tt, 256, 0, stream>>>(tmem, IDX, KV);
  // 4) entity embeddings and q/k/v projections (WMMA)
  k_make_qe<<<Bb * Oo, 256, 0, stream>>>(labels, entity_W, entity_b, QE);
  k_gemm_bf16<0, false, false><<<gemm_grid(Bb * Oo, Ee), 256, 0, stream>>>(
      QE, Wq, QBUF, nullptr, nullptr, Bb * Oo, Ee, Ee, Ee, Ee, Ee);
  k_gemm_bf16<0, false, false><<<gemm_grid(Bb * Nkv, Ee), 256, 0, stream>>>(
      KV, Wk, KBUF, nullptr, nullptr, Bb * Nkv, Ee, Dd, Dd, Ee, Ee);
  k_gemm_bf16<0, false, false><<<gemm_grid(Bb * Nkv, Ee), 256, 0, stream>>>(
      KV, Wv, VBUF, nullptr, nullptr, Bb * Nkv, Ee, Dd, Dd, Ee, Ee);
  // 5) attention
  k_scores<<<Bb * Hh * Oo, 128, 0, stream>>>(QBUF, KBUF, SCORES);
  k_attn<<<Bb * Hh * Oo, 128, 0, stream>>>(SCORES, VBUF, AOUT);
  k_qr<<<Bb * Oo, 64, 0, stream>>>(SCORES, ROWMAP);
  // 6) objectness MLP: gathered-row WMMA GEMM with tanh+bias epilogue
  k_gemm_bf16<1, false, true><<<gemm_grid(2 * NHI, Ee), 256, 0, stream>>>(
      KV, obj_W1, ZBUF, ROWMAP, obj_b1, 2 * NHI, Ee, Dd, Dd, Ee, Ee);
  k_mlp2<<<2 * NHI, 256, 0, stream>>>(ZBUF, obj_W2, obj_b2, SBUF);
  k_gce<<<1, 256, 0, stream>>>(SBUF, labels, GCE);
  // 7) output projection + broadcast gce
  k_gemm_bf16<0, false, false><<<gemm_grid(Bb * Oo, Ee), 256, 0, stream>>>(
      AOUT, Wo, FUSED, nullptr, nullptr, Bb * Oo, Ee, Ee, Ee, Ee, Ee);
  int ntot = Bb * Oo * Ee;
  k_final<<<(ntot + 255) / 256, 256, 0, stream>>>(FUSED, GCE, out, ntot);
}
